// CausalSelfAttention_56805237456959
// MI455X (gfx1250) — matrix operand order repaired and checked by reading, not compiled, not run on hardware
//
#include <hip/hip_runtime.h>
#include <hip/hip_bf16.h>

// ---------------- types ----------------
typedef __attribute__((ext_vector_type(16))) __bf16 bf16x16;
typedef __attribute__((ext_vector_type(8)))  __bf16 bf16x8;
typedef __attribute__((ext_vector_type(8)))  float  f32x8;

#define B_   2
#define T_   2048
#define DIM_ 2048
#define H_   16
#define D_   128
#define NQKV 6144
#define SCALE_Q 0.08838834764831845f  // 1/sqrt(128)

// ---------------- helpers ----------------
__device__ __forceinline__ __bf16 f2bf(float f) {
  unsigned u = __builtin_bit_cast(unsigned, f);
  u += 0x7fffu + ((u >> 16) & 1u);           // round-to-nearest-even
  unsigned short h = (unsigned short)(u >> 16);
  return __builtin_bit_cast(__bf16, h);
}
__device__ __forceinline__ float bf2f(__bf16 b) {
  unsigned short h = __builtin_bit_cast(unsigned short, b);
  unsigned u = ((unsigned)h) << 16;
  return __builtin_bit_cast(float, u);
}
__device__ __forceinline__ unsigned pack2(float a, float b) {
  unsigned short ha = __builtin_bit_cast(unsigned short, f2bf(a));
  unsigned short hb = __builtin_bit_cast(unsigned short, f2bf(b));
  return (unsigned)ha | ((unsigned)hb << 16);
}
// assemble a 16-element bf16 fragment from two contiguous 8-element chunks
__device__ __forceinline__ bf16x16 make_frag(const __bf16* p0, const __bf16* p1) {
  bf16x8 lo = *(const bf16x8*)p0;
  bf16x8 hi = *(const bf16x8*)p1;
  return __builtin_shufflevector(lo, hi, 0,1,2,3,4,5,6,7,8,9,10,11,12,13,14,15);
}
__device__ __forceinline__ f32x8 wmma_bf16(bf16x16 a, bf16x16 b, f32x8 c) {
  return __builtin_amdgcn_wmma_f32_16x16x32_bf16(false, a, false, b, (short)0, c,
                                                 false, false);
}
// CDNA5 async global->LDS copy of 32 bytes per lane (two B128 ops, ASYNCcnt).
__device__ __forceinline__ void async_copy32(const __bf16* gsrc, __bf16* ldst) {
  unsigned lds = (unsigned)(size_t)(void*)ldst;          // low 32 bits = LDS offset
  unsigned long long ga = (unsigned long long)(const void*)gsrc;
  asm volatile("global_load_async_to_lds_b128 %0, %1, off"
               :: "v"(lds), "v"(ga) : "memory");
  asm volatile("global_load_async_to_lds_b128 %0, %1, off offset:16"
               :: "v"(lds), "v"(ga) : "memory");
}
__device__ __forceinline__ void wait_async0() {
  asm volatile("s_wait_asynccnt 0x0" ::: "memory");
}

// =====================================================================
// Kernel 1: QKV projection.  qkv[m,n] = sum_k x[m,k] * Wqkv[n,k]
// 128x128 block tile, 8 waves (4 M x 2 N), wave tile 32x64, K-step 32.
// Epilogue: Q,K -> [B,H,T,D] bf16 ; V -> [B,H,D,T] bf16 (transposed).
// =====================================================================
__global__ __launch_bounds__(256) void qkv_gemm_kernel(
    const float* __restrict__ x, const float* __restrict__ Wqkv,
    __bf16* __restrict__ Qb, __bf16* __restrict__ Kb, __bf16* __restrict__ Vt) {
  __shared__ __bf16 sA[128][32];
  __shared__ __bf16 sW[128][32];

  const int tid  = threadIdx.x;
  const int lane = tid & 31;
  const int wv   = tid >> 5;          // 0..7
  const int mw   = wv & 3;            // 4 waves along M
  const int nw   = wv >> 2;           // 2 waves along N
  const int lo   = lane & 15;
  const int hi   = lane >> 4;
  const int mTile = blockIdx.y;       // 0..31
  const int nTile = blockIdx.x;       // 0..47

  f32x8 acc[2][4];
  for (int i = 0; i < 2; ++i)
    for (int j = 0; j < 4; ++j)
      acc[i][j] = (f32x8){0.f,0.f,0.f,0.f,0.f,0.f,0.f,0.f};

  const int ldRow = tid >> 1;          // 0..127
  const int ldCol = (tid & 1) * 16;    // 0 or 16
  const float* gA = x    + (size_t)(mTile * 128 + ldRow) * DIM_ + ldCol;
  const float* gW = Wqkv + (size_t)(nTile * 128 + ldRow) * DIM_ + ldCol;

  for (int k0 = 0; k0 < DIM_; k0 += 32) {
    __syncthreads();
    { // stage A (fp32 -> bf16)
      const float* g = gA + k0;
      float4 f0 = *(const float4*)(g + 0);
      float4 f1 = *(const float4*)(g + 4);
      float4 f2 = *(const float4*)(g + 8);
      float4 f3 = *(const float4*)(g + 12);
      unsigned* d = (unsigned*)&sA[ldRow][ldCol];
      d[0]=pack2(f0.x,f0.y); d[1]=pack2(f0.z,f0.w);
      d[2]=pack2(f1.x,f1.y); d[3]=pack2(f1.z,f1.w);
      d[4]=pack2(f2.x,f2.y); d[5]=pack2(f2.z,f2.w);
      d[6]=pack2(f3.x,f3.y); d[7]=pack2(f3.z,f3.w);
      if (k0 + 32 < DIM_) __builtin_prefetch(g + 32, 0, 1);
    }
    { // stage W (fp32 -> bf16)
      const float* g = gW + k0;
      float4 f0 = *(const float4*)(g + 0);
      float4 f1 = *(const float4*)(g + 4);
      float4 f2 = *(const float4*)(g + 8);
      float4 f3 = *(const float4*)(g + 12);
      unsigned* d = (unsigned*)&sW[ldRow][ldCol];
      d[0]=pack2(f0.x,f0.y); d[1]=pack2(f0.z,f0.w);
      d[2]=pack2(f1.x,f1.y); d[3]=pack2(f1.z,f1.w);
      d[4]=pack2(f2.x,f2.y); d[5]=pack2(f2.z,f2.w);
      d[6]=pack2(f3.x,f3.y); d[7]=pack2(f3.z,f3.w);
      if (k0 + 32 < DIM_) __builtin_prefetch(g + 32, 0, 1);
    }
    __syncthreads();

    bf16x16 afr[2];
    for (int mf = 0; mf < 2; ++mf) {
      int r = mw * 32 + mf * 16 + lo;
      int kp = hi * 8;
      afr[mf] = make_frag(&sA[r][kp], &sA[r][kp + 16]);
    }
    bf16x16 bfr[4];
    for (int nf = 0; nf < 4; ++nf) {
      int n = nw * 64 + nf * 16 + lo;
      int kb = hi * 8;
      bfr[nf] = make_frag(&sW[n][kb], &sW[n][kb + 16]);
    }
    for (int mf = 0; mf < 2; ++mf)
      for (int nf = 0; nf < 4; ++nf)
        acc[mf][nf] = wmma_bf16(afr[mf], bfr[nf], acc[mf][nf]);
  }

  // epilogue: destination (Q/K/V) is uniform per block -> scalar branch
  const int sel = nTile >> 4;              // 0..2
  const int nInner = (nTile & 15) * 128;   // col offset inside the 2048 slab
  for (int mf = 0; mf < 2; ++mf)
    for (int nf = 0; nf < 4; ++nf)
      for (int r = 0; r < 8; ++r) {
        int m = mTile * 128 + mw * 32 + mf * 16 + r + 8 * hi;
        int idx = nInner + nw * 64 + nf * 16 + lo;
        int bi = m >> 11;                  // / T_
        int t  = m & (T_ - 1);
        int h  = idx >> 7;
        int d  = idx & 127;
        __bf16 v = f2bf(acc[mf][nf][r]);
        size_t bh = (size_t)(bi * H_ + h);
        if (sel == 0)      Qb[(bh * T_ + t) * D_ + d] = v;
        else if (sel == 1) Kb[(bh * T_ + t) * D_ + d] = v;
        else               Vt[(bh * D_ + d) * T_ + t] = v;
      }
}

// =====================================================================
// Kernel 2: RoPE in place on [B,H,T,D] bf16 (scale folds 1/sqrt(D) for Q)
// =====================================================================
__global__ __launch_bounds__(256) void rope_kernel(__bf16* __restrict__ p, float scale) {
  size_t idx = (size_t)blockIdx.x * 256 + threadIdx.x;   // B*H*T*64 threads
  int d = (int)(idx & 63);
  size_t rest = idx >> 6;
  int t = (int)(rest & (T_ - 1));
  size_t bh = rest >> 11;
  __bf16* row = p + (bh * T_ + t) * D_;
  float invf = __powf(10000.0f, -(float)(2 * d) / (float)D_);
  float th = (float)t * invf;
  float s, c;
  __sincosf(th, &s, &c);
  float x1 = bf2f(row[d]);
  float x2 = bf2f(row[d + 64]);
  row[d]      = f2bf((x1 * c - x2 * s) * scale);
  row[d + 64] = f2bf((x2 * c + x1 * s) * scale);
}

// =====================================================================
// Kernel 3: causal flash attention.
// grid = (T/128, B*H).  8 waves/block, wave owns 16 q rows.
// K / V^T staged via global_load_async_to_lds_b128 (ASYNCcnt);
// online softmax in f32; P transposed through per-wave LDS.
// =====================================================================
__global__ __launch_bounds__(256) void flash_kernel(
    const __bf16* __restrict__ Qb, const __bf16* __restrict__ Kb,
    const __bf16* __restrict__ Vt, __bf16* __restrict__ Ao) {
  __shared__ __bf16 sK[32][128];
  __shared__ __bf16 sV[128][32];
  __shared__ __bf16 sP[8][16][32];

  const int tid  = threadIdx.x;
  const int lane = tid & 31;
  const int wv   = tid >> 5;
  const int lo   = lane & 15;
  const int hi   = lane >> 4;
  const int qTile = blockIdx.x;          // 0..15
  const int bh    = blockIdx.y;          // 0..31
  const int qRowBase = qTile * 128 + wv * 16;

  // Q fragments, loaded straight from global in A-fragment layout
  const __bf16* qrow = Qb + ((size_t)bh * T_ + qRowBase + lo) * D_;
  bf16x16 qf[4];
  for (int dc = 0; dc < 4; ++dc) {
    int kp = hi * 8;
    qf[dc] = make_frag(qrow + dc * 32 + kp, qrow + dc * 32 + kp + 16);
  }

  f32x8 accO[8];
  for (int f = 0; f < 8; ++f) accO[f] = (f32x8){0.f,0.f,0.f,0.f,0.f,0.f,0.f,0.f};
  float mrow[8], lrow[8];
  for (int r = 0; r < 8; ++r) { mrow[r] = -1e30f; lrow[r] = 0.f; }

  const int kr  = tid >> 3, dcb = (tid & 7) * 16;      // K-tile copy mapping
  const int vr  = tid >> 1, vcb = (tid & 1) * 16;      // V-tile copy mapping
  const __bf16* gK = Kb + ((size_t)bh * T_ + kr) * D_ + dcb;
  const __bf16* gV = Vt + ((size_t)bh * D_ + vr) * T_ + vcb;

  const int numKB = (qTile + 1) * 4;     // key blocks of 32 keys
  for (int kb = 0; kb < numKB; ++kb) {
    const int kbase = kb * 32;
    __syncthreads();
    // stage K block [32 kj][128 d] and V^T block [128 d][32 kj] via TDM-lite
    async_copy32(gK + (size_t)kbase * D_, &sK[kr][dcb]);
    async_copy32(gV + kbase,              &sV[vr][vcb]);
    wait_async0();
    __syncthreads();

    // S = Q . K^T   (16 q x 32 k, split into two 16x16 accumulators)
    f32x8 s0 = (f32x8){0.f,0.f,0.f,0.f,0.f,0.f,0.f,0.f};
    f32x8 s1 = s0;
    for (int dc = 0; dc < 4; ++dc) {
      bf16x16 bk0 = make_frag(&sK[lo][dc * 32 + hi * 8],
                              &sK[lo][dc * 32 + hi * 8 + 16]);
      bf16x16 bk1 = make_frag(&sK[16 + lo][dc * 32 + hi * 8],
                              &sK[16 + lo][dc * 32 + hi * 8 + 16]);
      s0 = wmma_bf16(qf[dc], bk0, s0);
      s1 = wmma_bf16(qf[dc], bk1, s1);
    }

    // causal mask + online softmax (row reductions within 16-lane half-groups)
    for (int r = 0; r < 8; ++r) {
      int qi  = qRowBase + r + 8 * hi;
      float v0 = s0[r]; if (kbase + lo > qi)      v0 = -1e30f;
      float v1 = s1[r]; if (kbase + 16 + lo > qi) v1 = -1e30f;
      float mx = fmaxf(v0, v1);
      for (int m = 1; m < 16; m <<= 1) mx = fmaxf(mx, __shfl_xor(mx, m, 32));
      float mnew = fmaxf(mrow[r], mx);
      float p0 = __expf(v0 - mnew);
      float p1 = __expf(v1 - mnew);
      float ps = p0 + p1;
      for (int m = 1; m < 16; m <<= 1) ps += __shfl_xor(ps, m, 32);
      float fac = __expf(mrow[r] - mnew);
      lrow[r] = lrow[r] * fac + ps;
      mrow[r] = mnew;
      for (int f = 0; f < 8; ++f) accO[f][r] *= fac;
      int prow = r + 8 * hi;
      sP[wv][prow][lo]      = f2bf(p0);
      sP[wv][prow][16 + lo] = f2bf(p1);
    }
    // order same-wave LDS RAW (P store -> P fragment load)
    asm volatile("s_wait_dscnt 0x0" ::: "memory");

    bf16x16 pf = make_frag(&sP[wv][lo][hi * 8], &sP[wv][lo][hi * 8 + 16]);
    for (int f = 0; f < 8; ++f) {
      bf16x16 bv = make_frag(&sV[f * 16 + lo][hi * 8],
                             &sV[f * 16 + lo][hi * 8 + 16]);
      accO[f] = wmma_bf16(pf, bv, accO[f]);
    }
  }

  // normalize + store attn output as bf16 [B*T, H*D]
  const int bi = bh >> 4;
  const int h  = bh & 15;
  for (int f = 0; f < 8; ++f)
    for (int r = 0; r < 8; ++r) {
      int qi = qRowBase + r + 8 * hi;
      int d  = f * 16 + lo;
      float v = accO[f][r] / lrow[r];
      Ao[((size_t)(bi * T_ + qi)) * (H_ * D_) + h * D_ + d] = f2bf(v);
    }
}

// =====================================================================
// Kernel 4: output projection.  out[m,o] = sum_i Ao[m,i] * Wp[o,i]
// A is bf16 (workspace, async-staged), W fp32 -> bf16, output fp32.
// =====================================================================
__global__ __launch_bounds__(256) void proj_gemm_kernel(
    const __bf16* __restrict__ Ao, const float* __restrict__ Wp,
    float* __restrict__ out) {
  __shared__ __bf16 sA[128][32];
  __shared__ __bf16 sW[128][32];

  const int tid  = threadIdx.x;
  const int lane = tid & 31;
  const int wv   = tid >> 5;
  const int mw   = wv & 3;
  const int nw   = wv >> 2;
  const int lo   = lane & 15;
  const int hi   = lane >> 4;
  const int mTile = blockIdx.y;   // 0..31
  const int nTile = blockIdx.x;   // 0..15

  f32x8 acc[2][4];
  for (int i = 0; i < 2; ++i)
    for (int j = 0; j < 4; ++j)
      acc[i][j] = (f32x8){0.f,0.f,0.f,0.f,0.f,0.f,0.f,0.f};

  const int ldRow = tid >> 1;
  const int ldCol = (tid & 1) * 16;
  const __bf16* gA = Ao + (size_t)(mTile * 128 + ldRow) * DIM_ + ldCol;
  const float*  gW = Wp + (size_t)(nTile * 128 + ldRow) * DIM_ + ldCol;

  for (int k0 = 0; k0 < DIM_; k0 += 32) {
    __syncthreads();
    // stage A (bf16): async global -> LDS, no VGPR round-trip
    async_copy32(gA + k0, &sA[ldRow][ldCol]);
    { // stage W (fp32 -> bf16)
      const float* g = gW + k0;
      float4 f0 = *(const float4*)(g + 0);
      float4 f1 = *(const float4*)(g + 4);
      float4 f2 = *(const float4*)(g + 8);
      float4 f3 = *(const float4*)(g + 12);
      unsigned* d = (unsigned*)&sW[ldRow][ldCol];
      d[0]=pack2(f0.x,f0.y); d[1]=pack2(f0.z,f0.w);
      d[2]=pack2(f1.x,f1.y); d[3]=pack2(f1.z,f1.w);
      d[4]=pack2(f2.x,f2.y); d[5]=pack2(f2.z,f2.w);
      d[6]=pack2(f3.x,f3.y); d[7]=pack2(f3.z,f3.w);
      if (k0 + 32 < DIM_) __builtin_prefetch(g + 32, 0, 1);
    }
    wait_async0();
    __syncthreads();

    bf16x16 afr[2];
    for (int mf = 0; mf < 2; ++mf) {
      int r = mw * 32 + mf * 16 + lo;
      int kp = hi * 8;
      afr[mf] = make_frag(&sA[r][kp], &sA[r][kp + 16]);
    }
    bf16x16 bfr[4];
    for (int nf = 0; nf < 4; ++nf) {
      int n = nw * 64 + nf * 16 + lo;
      int kb = hi * 8;
      bfr[nf] = make_frag(&sW[n][kb], &sW[n][kb + 16]);
    }
    for (int mf = 0; mf < 2; ++mf)
      for (int nf = 0; nf < 4; ++nf)
        acc[mf][nf] = wmma_bf16(afr[mf], bfr[nf], acc[mf][nf]);
  }

  for (int mf = 0; mf < 2; ++mf)
    for (int nf = 0; nf < 4; ++nf)
      for (int r = 0; r < 8; ++r) {
        int m = mTile * 128 + mw * 32 + mf * 16 + r + 8 * hi;
        int n = nTile * 128 + nw * 64 + nf * 16 + lo;
        out[(size_t)m * DIM_ + n] = acc[mf][nf][r];
      }
}

// =====================================================================
extern "C" void kernel_launch(void* const* d_in, const int* in_sizes, int n_in,
                              void* d_out, int out_size, void* d_ws, size_t ws_size,
                              hipStream_t stream) {
  const float* x     = (const float*)d_in[0];   // [B,T,2048]
  const float* W_qkv = (const float*)d_in[1];   // [6144,2048]
  const float* W_prj = (const float*)d_in[2];   // [2048,2048]
  float* out = (float*)d_out;                   // [B,T,2048]

  const size_t nElem = (size_t)B_ * H_ * T_ * D_;      // 8,388,608
  char* ws = (char*)d_ws;
  __bf16* Qb = (__bf16*)(ws);
  __bf16* Kb = (__bf16*)(ws + 1 * nElem * sizeof(__bf16));
  __bf16* Vt = (__bf16*)(ws + 2 * nElem * sizeof(__bf16));
  __bf16* Ao = (__bf16*)(ws + 3 * nElem * sizeof(__bf16));

  // 1) QKV projection -> Q,K [B,H,T,D], V^T [B,H,D,T]  (bf16)
  qkv_gemm_kernel<<<dim3(NQKV / 128, (B_ * T_) / 128), 256, 0, stream>>>(
      x, W_qkv, Qb, Kb, Vt);

  // 2) RoPE (Q also gets 1/sqrt(D) folded in)
  const int ropeBlocks = (int)((size_t)B_ * H_ * T_ * 64 / 256);
  rope_kernel<<<ropeBlocks, 256, 0, stream>>>(Qb, SCALE_Q);
  rope_kernel<<<ropeBlocks, 256, 0, stream>>>(Kb, 1.0f);

  // 3) causal flash attention -> Ao [B*T, H*D] bf16
  flash_kernel<<<dim3(T_ / 128, B_ * H_), 256, 0, stream>>>(Qb, Kb, Vt, Ao);

  // 4) output projection -> fp32
  proj_gemm_kernel<<<dim3(DIM_ / 128, (B_ * T_) / 128), 256, 0, stream>>>(
      Ao, W_prj, out);
}
